// DynamicallyQuantizedLinear_40913858461871
// MI455X (gfx1250) — 8-bit matrix operand order repaired and checked by reading, compiled, not run on hardware
//
#include <hip/hip_runtime.h>
#include <stdint.h>

typedef __attribute__((ext_vector_type(8))) int      v8i;
typedef __attribute__((ext_vector_type(4))) float    v4f;
typedef __attribute__((ext_vector_type(4))) unsigned v4u;

constexpr int Mdim = 32768;
constexpr int Kdim = 1024;
constexpr int Ndim = 1024;
constexpr int KB   = Kdim / 64;
constexpr int MT   = Mdim / 16;
constexpr int NT   = Ndim / 16;

__device__ __forceinline__ unsigned umin2(unsigned a, unsigned b) { return a < b ? a : b; }
__device__ __forceinline__ unsigned umax2(unsigned a, unsigned b) { return a > b ? a : b; }

__device__ __forceinline__ unsigned fkey(float f) {
  unsigned u = __float_as_uint(f);
  return u ^ (unsigned)(((int)u >> 31) | (int)0x80000000);
}
__device__ __forceinline__ float fdec(unsigned k) {
  unsigned u = (k & 0x80000000u) ? (k ^ 0x80000000u) : ~k;
  return __uint_as_float(u);
}

__global__ void k_init(unsigned* keys) {
  keys[0] = 0xFFFFFFFFu;
  keys[1] = 0u;
}

__global__ __launch_bounds__(256) void k_minmax(const float* __restrict__ x,
                                                unsigned* __restrict__ keys, int n4) {
  __shared__ unsigned smin[8], smax[8];
  unsigned kmin = 0xFFFFFFFFu, kmax = 0u;
  const v4f* x4 = (const v4f*)x;
  for (int i = blockIdx.x * blockDim.x + threadIdx.x; i < n4; i += gridDim.x * blockDim.x) {
    v4f v = x4[i];
#pragma unroll
    for (int j = 0; j < 4; ++j) {
      unsigned kk = fkey(v[j]);
      kmin = umin2(kmin, kk);
      kmax = umax2(kmax, kk);
    }
  }
#pragma unroll
  for (int off = 16; off >= 1; off >>= 1) {
    kmin = umin2(kmin, (unsigned)__shfl_xor((int)kmin, off, 32));
    kmax = umax2(kmax, (unsigned)__shfl_xor((int)kmax, off, 32));
  }
  int lane = threadIdx.x & 31, wave = threadIdx.x >> 5;
  if (lane == 0) { smin[wave] = kmin; smax[wave] = kmax; }
  __syncthreads();
  if (threadIdx.x == 0) {
    unsigned bmin = smin[0], bmax = smax[0];
#pragma unroll
    for (int w = 1; w < 8; ++w) { bmin = umin2(bmin, smin[w]); bmax = umax2(bmax, smax[w]); }
    atomicMin(&keys[0], bmin);
    atomicMax(&keys[1], bmax);
  }
}

__global__ void k_scale(const unsigned* __restrict__ keys, float* __restrict__ sc) {
  float mn = fminf(fdec(keys[0]), 0.0f);
  float mx = fmaxf(fdec(keys[1]), 0.0f);
  float s  = fmaxf((mx - mn) * (1.0f / 255.0f), 1.1920928955078125e-07f);
  float z  = fminf(127.0f, fmaxf(-128.0f, -128.0f - rintf(mn / s)));
  sc[0] = s;
  sc[1] = z;
}

__global__ __launch_bounds__(256) void k_quant_pack(const float* __restrict__ x,
                                                    const float* __restrict__ sc,
                                                    unsigned* __restrict__ apk) {
  int t    = blockIdx.x * blockDim.x + threadIdx.x;
  int lane = t & 31;
  int kb   = (t >> 5) & (KB - 1);
  int mt   = t >> 9;
  int m    = mt * 16 + (lane & 15);
  int hi   = lane >> 4;
  float s  = sc[0], z = sc[1];
  float inv = 1.0f / s;
  const float* xr = x + (size_t)m * Kdim + kb * 64 + hi * 8;
  unsigned d[8];
#pragma unroll
  for (int v = 0; v < 8; ++v) {
    int koff = (v & 1) * 4 + (v >> 1) * 16;
    v4f f = *(const v4f*)(xr + koff);
    unsigned dw = 0;
#pragma unroll
    for (int j = 0; j < 4; ++j) {
      float qf = fminf(127.0f, fmaxf(-128.0f, rintf(f[j] * inv) + z));
      dw |= ((unsigned)((int)qf & 0xFF)) << (8 * j);
    }
    d[v] = dw;
  }
  unsigned base = (unsigned)(((mt * KB + kb) * 2) * 32 + lane) * 4;
  *(v4u*)(apk + base)        = (v4u){d[0], d[1], d[2], d[3]};
  *(v4u*)(apk + base + 128)  = (v4u){d[4], d[5], d[6], d[7]};
}

__global__ __launch_bounds__(256) void k_repack_w(const signed char* __restrict__ w,
                                                  unsigned* __restrict__ wpk) {
  int t    = blockIdx.x * blockDim.x + threadIdx.x;
  int lane = t & 31;
  int kb   = (t >> 5) & (KB - 1);
  int nt   = t >> 9;
  int n    = nt * 16 + (lane & 15);
  int hi   = lane >> 4;
  unsigned d[8];
#pragma unroll
  for (int v = 0; v < 8; ++v) {
    int k0 = kb * 64 + hi * 8 + (v & 1) * 4 + (v >> 1) * 16;
    unsigned dw = 0;
#pragma unroll
    for (int j = 0; j < 4; ++j)
      dw |= ((unsigned)(unsigned char)w[(size_t)(k0 + j) * Ndim + n]) << (8 * j);
    d[v] = dw;
  }
  unsigned base = (unsigned)(((nt * KB + kb) * 2) * 32 + lane) * 4;
  *(v4u*)(wpk + base)       = (v4u){d[0], d[1], d[2], d[3]};
  *(v4u*)(wpk + base + 128) = (v4u){d[4], d[5], d[6], d[7]};
}

__global__ __launch_bounds__(256) void k_gemm(const unsigned* __restrict__ apk,
                                              const unsigned* __restrict__ wpk,
                                              const float* __restrict__ sc,
                                              const float* __restrict__ wscale,
                                              const int* __restrict__ wsum,
                                              const float* __restrict__ bias,
                                              float* __restrict__ out) {
  int lane = threadIdx.x & 31;
  int wave = threadIdx.x >> 5;
  int wm   = wave >> 2;
  int wn   = wave & 3;
  int mt0  = blockIdx.y * 8 + wm * 4;
  int nt0  = blockIdx.x * 8 + wn * 2;

  const v8i vzero = {0, 0, 0, 0, 0, 0, 0, 0};
  v8i acc[4][2];
#pragma unroll
  for (int i = 0; i < 4; ++i)
#pragma unroll
    for (int j = 0; j < 2; ++j) acc[i][j] = vzero;

  for (int kb = 0; kb < KB; ++kb) {
    v8i a[4], b[2];
#pragma unroll
    for (int i = 0; i < 4; ++i) {
      const unsigned* p = apk + (size_t)(((mt0 + i) * KB + kb) * 64 + lane) * 4;
      v4u lo = *(const v4u*)p;
      v4u hi = *(const v4u*)(p + 128);
      v8i av;
      av[0] = lo[0]; av[1] = lo[1]; av[2] = lo[2]; av[3] = lo[3];
      av[4] = hi[0]; av[5] = hi[1]; av[6] = hi[2]; av[7] = hi[3];
      a[i] = av;
    }
#pragma unroll
    for (int j = 0; j < 2; ++j) {
      const unsigned* p = wpk + (size_t)(((nt0 + j) * KB + kb) * 64 + lane) * 4;
      v4u lo = *(const v4u*)p;
      v4u hi = *(const v4u*)(p + 128);
      v8i bv;
      bv[0] = lo[0]; bv[1] = lo[1]; bv[2] = lo[2]; bv[3] = lo[3];
      bv[4] = hi[0]; bv[5] = hi[1]; bv[6] = hi[2]; bv[7] = hi[3];
      b[j] = bv;
    }
    if (kb + 1 < KB) {
      __builtin_prefetch(apk + (size_t)((mt0 * KB + kb + 1) * 64 + lane) * 4, 0, 1);
    }
#pragma unroll
    for (int i = 0; i < 4; ++i)
#pragma unroll
      for (int j = 0; j < 2; ++j)
        acc[i][j] = __builtin_amdgcn_wmma_i32_16x16x64_iu8(
             true, a[i],  true, b[j], acc[i][j],
             false,  false);
  }

  float s = sc[0], z = sc[1];
  int col = lane & 15, hi = lane >> 4;
#pragma unroll
  for (int j = 0; j < 2; ++j) {
    int n      = (nt0 + j) * 16 + col;
    float wsc  = s * wscale[n];
    float corr = z * (float)wsum[n];
    float bb   = bias[n];
#pragma unroll
    for (int i = 0; i < 4; ++i) {
      int mbase = (mt0 + i) * 16 + hi * 8;
#pragma unroll
      for (int r = 0; r < 8; ++r) {
        out[(size_t)(mbase + r) * Ndim + n] = wsc * ((float)acc[i][j][r] - corr) + bb;
      }
    }
  }
}

extern "C" void kernel_launch(void* const* d_in, const int* in_sizes, int n_in,
                              void* d_out, int out_size, void* d_ws, size_t ws_size,
                              hipStream_t stream) {
  const float*       x      = (const float*)d_in[0];
  const signed char* w      = (const signed char*)d_in[1];
  const float*       wscale = (const float*)d_in[2];
  const int*         wsum   = (const int*)d_in[3];
  const float*       bias   = (const float*)d_in[4];
  float*             out    = (float*)d_out;

  char*     ws   = (char*)d_ws;
  unsigned* keys = (unsigned*)ws;
  float*    sc   = (float*)(ws + 8);
  unsigned* wpk  = (unsigned*)(ws + (1u << 20));
  unsigned* apk  = (unsigned*)(ws + (2u << 20));

  k_init<<<1, 1, 0, stream>>>(keys);
  k_minmax<<<1024, 256, 0, stream>>>(x, keys, (Mdim * Kdim) / 4);
  k_scale<<<1, 1, 0, stream>>>(keys, sc);
  k_quant_pack<<<(MT * KB * 32) / 256, 256, 0, stream>>>(x, sc, apk);
  k_repack_w<<<(NT * KB * 32) / 256, 256, 0, stream>>>(w, wpk);
  dim3 grid(Ndim / 128, Mdim / 128);
  k_gemm<<<grid, 256, 0, stream>>>(apk, wpk, sc, wscale, wsum, bias, out);
}
